// SparsePropConv_49306224558792
// MI455X (gfx1250) — matrix operand order repaired and checked by reading, not compiled, not run on hardware
//
#include <hip/hip_runtime.h>
#include <hip/hip_bf16.h>

// ---------------------------------------------------------------------------
// SparsePropConv for MI455X (gfx1250, wave32, WMMA).
// Chain of C=512 -> C=512 convs (k=1 / k=2) + 2 maxpools, each layer's output
// scattered onto a strided diagonal of a [64,512,64,64] fp32 map.
// GEMM path: bf16 activations/weights, v_wmma_f32_16x16x32_bf16, f32 accum.
//  - Activations channels-last act[b][l][c]: B-fragment = one 32B load/lane.
//  - Block stages its 64-row weight tile (all ci, all taps) in LDS once,
//    padded row stride (520 bf16) for conflict-free 16B fragment reads;
//    removes the 4x redundant global A-fetch across the block's N-waves.
//  - K loop fully unrolled: 64 (k=1) / 128 (k=2) WMMAs per wave.
// ---------------------------------------------------------------------------

typedef __attribute__((ext_vector_type(16))) __bf16 v16bf;
typedef __attribute__((ext_vector_type(8)))  __bf16 v8bf;
typedef __attribute__((ext_vector_type(8)))  float  v8f;
typedef __attribute__((ext_vector_type(4)))  float  v4f;

__device__ __forceinline__ __bf16 f2bf(float f) {
    union { float f; unsigned u; } in; in.f = f;
    unsigned u = in.u;
    unsigned r = (u + 0x7FFFu + ((u >> 16) & 1u)) >> 16;   // round-nearest-even
    union { unsigned short s; __bf16 b; } out; out.s = (unsigned short)r;
    return out.b;
}
__device__ __forceinline__ float bf2f(__bf16 b) {
    union { unsigned short s; __bf16 b; } in; in.b = b;
    union { unsigned u; float f; } out; out.u = ((unsigned)in.s) << 16;
    return out.f;
}

// --------------------------- utility kernels -------------------------------

__global__ __launch_bounds__(256) void zero_f4(v4f* p, size_t n4) {
    size_t i = (size_t)blockIdx.x * blockDim.x + threadIdx.x;
    size_t stp = (size_t)gridDim.x * blockDim.x;
    v4f z = (v4f)(0.0f);
    for (; i < n4; i += stp) p[i] = z;
}

__global__ void mask_kernel(float* mask) {
    for (int layer = 0; layer < 32; ++layer) {
        int stride, offset, len;
        if (layer < 16)      { stride = 1; offset = layer;            len = 64 - layer; }
        else if (layer < 24) { int o = layer - 16; stride = 2; offset = 17 + 2*o; len = 24 - o; }
        else                 { int o = layer - 24; stride = 4; offset = 35 + 4*o; len = 8 - o; }
        for (int t = threadIdx.x; t < len; t += blockDim.x) {
            int i = t * stride, j = offset + t * stride;
            mask[i * 64 + j] = 1.0f;
        }
    }
}

// x[b][c][l] f32 -> act[b][l][c] bf16  (B=64, C=512, L=64)
__global__ __launch_bounds__(256) void cvt_x(const float* __restrict__ x,
                                             __bf16* __restrict__ act) {
    const size_t n = (size_t)64 * 64 * 512;
    size_t i = (size_t)blockIdx.x * blockDim.x + threadIdx.x;
    size_t stp = (size_t)gridDim.x * blockDim.x;
    for (; i < n; i += stp) {
        int c = (int)(i & 511);
        int l = (int)((i >> 9) & 63);
        int b = (int)(i >> 15);
        act[i] = f2bf(x[((size_t)b * 512 + c) * 64 + l]);
    }
}

// w1[co][ci][1] f32 -> wb1[co][ci] bf16
__global__ __launch_bounds__(256) void cvt_w1(const float* __restrict__ w,
                                              __bf16* __restrict__ o) {
    const size_t n = (size_t)512 * 512;
    size_t i = (size_t)blockIdx.x * blockDim.x + threadIdx.x;
    size_t stp = (size_t)gridDim.x * blockDim.x;
    for (; i < n; i += stp) o[i] = f2bf(w[i]);
}

// w2[k][co][ci][t] f32 -> wb2[k][t][co][ci] bf16   (k = 29 layers, t = 2 taps)
__global__ __launch_bounds__(256) void cvt_w2(const float* __restrict__ w,
                                              __bf16* __restrict__ o) {
    const size_t n = (size_t)29 * 2 * 512 * 512;
    size_t i = (size_t)blockIdx.x * blockDim.x + threadIdx.x;
    size_t stp = (size_t)gridDim.x * blockDim.x;
    for (; i < n; i += stp) {
        int ci = (int)(i & 511);
        int co = (int)((i >> 9) & 511);
        int t  = (int)((i >> 18) & 1);
        int k  = (int)(i >> 19);
        o[i] = f2bf(w[((((size_t)k * 512 + co) * 512) + ci) * 2 + t]);
    }
}

// maxpool(k=3, s=2) over l, channels-last, + relu + diagonal scatter.
__global__ __launch_bounds__(256) void maxpool_scatter(
    const __bf16* __restrict__ in, __bf16* __restrict__ out,
    float* __restrict__ x2d, int Lin, int Lout, int dstr, int doff) {
    const size_t n = (size_t)64 * Lout * 512;
    size_t idx = (size_t)blockIdx.x * blockDim.x + threadIdx.x;
    if (idx >= n) return;
    int c = (int)(idx & 511);
    size_t rest = idx >> 9;
    int l = (int)(rest % (size_t)Lout);
    int b = (int)(rest / (size_t)Lout);
    const __bf16* p = in + ((size_t)b * Lin + 2 * l) * 512 + c;
    float a0 = bf2f(p[0]);
    float a1 = bf2f(p[512]);
    float a2 = bf2f(p[1024]);
    float m = fmaxf(a0, fmaxf(a1, a2));
    m = fmaxf(m, 0.0f);  // relu (no-op on nonneg chain, kept for fidelity)
    out[idx] = f2bf(m);
    int i = l * dstr, j = doff + l * dstr;
    x2d[(((size_t)b * 512 + c) * 64 + i) * 64 + j] = m;
}

// --------------------------- WMMA conv GEMM --------------------------------
// Y[co, (b,l)] = relu( sum_t W_t[co,:] . act[b, l+t, :] + bias[co] )
// Block: 256 thr = 8 waves, wave tile 32x32 (2x2 of 16x16), block tile 64x128.
// LDS: full 64-row weight tile, padded row stride for bank-conflict-free reads.

union ABfrag { v16bf v; v8bf h[2]; };

#define LDSROW 520   // 512 ci + 8 pad (bf16 elems) -> 1040B row stride

template<int TAPS>
__global__ __launch_bounds__(256) void conv_wmma(
    const __bf16* __restrict__ actIn, __bf16* __restrict__ actOut,
    const __bf16* __restrict__ W, const float* __restrict__ bias,
    float* __restrict__ x2d, int Lin, int Lout, int dstr, int doff)
{
    constexpr int C = 512;
    extern __shared__ __bf16 smem[];    // TAPS*64*LDSROW bf16 (66.5/133 KB)

    const int lane = threadIdx.x & 31;
    const int wv   = threadIdx.x >> 5;
    const int wm   = wv >> 2;        // 0..1  (M waves)
    const int wn   = wv & 3;         // 0..3  (N waves)
    const int rowt = lane & 15;
    const int half = lane >> 4;

    const int co_blk = blockIdx.x * 64;
    const int n0     = blockIdx.y * 128 + wn * 32;
    const int Ncols  = 64 * Lout;

    // ---- stage block weight tile into LDS (all threads; coalesced 16B) ----
    {
        const int total8 = TAPS * 64 * (C / 8);       // v8bf chunks
        for (int idx = threadIdx.x; idx < total8; idx += 256) {
            int t   = idx >> 12;                      // idx / 4096 (tap)
            int rem = idx & 4095;
            int row = rem >> 6;                       // 0..63 (co within block)
            int c8  = (rem & 63) * 8;                 // ci start
            v8bf v = *(const v8bf*)(W + ((size_t)t * C + (co_blk + row)) * C + c8);
            *(v8bf*)(smem + (t * 64 + row) * LDSROW + c8) = v;
        }
    }
    __syncthreads();
    if (n0 >= Ncols) return;     // wave-uniform: Ncols is a multiple of 32

    // Column mapping for the two 16-wide N tiles this wave owns.
    int bIdx[2], lIdx[2];
    const __bf16* bBase[2];
#pragma unroll
    for (int nt = 0; nt < 2; ++nt) {
        int col = n0 + nt * 16 + rowt;
        int b = col / Lout;
        int l = col - b * Lout;
        bIdx[nt] = b; lIdx[nt] = l;
        bBase[nt] = actIn + ((size_t)b * Lin + l) * C + half * 16;
    }
    // A (weight) fragment offsets in LDS, per M tile and tap.
    int aOff[2][TAPS];
#pragma unroll
    for (int mt = 0; mt < 2; ++mt)
#pragma unroll
        for (int t = 0; t < TAPS; ++t)
            aOff[mt][t] = (t * 64 + wm * 32 + mt * 16 + rowt) * LDSROW + half * 16;

    v8f acc[2][2];
#pragma unroll
    for (int mt = 0; mt < 2; ++mt)
#pragma unroll
        for (int nt = 0; nt < 2; ++nt)
            acc[mt][nt] = (v8f)(0.0f);

#pragma unroll
    for (int kb = 0; kb < C; kb += 32) {
#pragma unroll
        for (int t = 0; t < TAPS; ++t) {
            ABfrag a[2];
#pragma unroll
            for (int mt = 0; mt < 2; ++mt) {
                a[mt].h[0] = *(const v8bf*)(smem + aOff[mt][t] + kb);      // K=kb+16h..
                a[mt].h[1] = *(const v8bf*)(smem + aOff[mt][t] + kb + 8);  // K=kb+16h+8..
            }
            v16bf bf[2];
#pragma unroll
            for (int nt = 0; nt < 2; ++nt)
                bf[nt] = *(const v16bf*)(bBase[nt] + (size_t)t * C + kb);  // K=kb+16h..
#pragma unroll
            for (int mt = 0; mt < 2; ++mt)
#pragma unroll
                for (int nt = 0; nt < 2; ++nt)
                    acc[mt][nt] = __builtin_amdgcn_wmma_f32_16x16x32_bf16(
                        false, a[mt].v, false, bf[nt],
                        (short)0, acc[mt][nt], false, false);
        }
    }

    // Epilogue: bias + relu, bf16 store to next activation, fp32 diagonal scatter.
#pragma unroll
    for (int mt = 0; mt < 2; ++mt) {
#pragma unroll
        for (int nt = 0; nt < 2; ++nt) {
            const int cob = co_blk + wm * 32 + mt * 16 + half * 8; // 8 consecutive rows
            const int b = bIdx[nt], l = lIdx[nt];
            const int i = l * dstr, j = doff + l * dstr;
            float* xp = x2d + (((size_t)b * C + cob) * 64 + i) * 64 + j;
            v8bf pk;
#pragma unroll
            for (int r = 0; r < 8; ++r) {
                float f = acc[mt][nt][r] + bias[cob + r];
                f = f > 0.0f ? f : 0.0f;
                xp[(size_t)r * 4096] = f;               // co stride = 64*64
                pk[r] = f2bf(f);
            }
            *(v8bf*)(actOut + ((size_t)b * Lout + l) * C + cob) = pk;
        }
    }
}

// ------------------------------- driver ------------------------------------

extern "C" void kernel_launch(void* const* d_in, const int* in_sizes, int n_in,
                              void* d_out, int out_size, void* d_ws, size_t ws_size,
                              hipStream_t stream) {
    const float* x  = (const float*)d_in[0];
    const float* w1 = (const float*)d_in[1];
    const float* b1 = (const float*)d_in[2];
    const float* w2 = (const float*)d_in[3];
    const float* b2 = (const float*)d_in[4];

    float* out  = (float*)d_out;
    float* x2d  = out;
    float* mask = out + (size_t)64 * 512 * 64 * 64;

    // Workspace carving (~39 MB): two ping-pong activation buffers + bf16 weights.
    char* ws = (char*)d_ws;
    __bf16* actA = (__bf16*)(ws);                               // 4 MB
    __bf16* actB = (__bf16*)(ws + ((size_t)4  << 20));          // 4 MB
    __bf16* wb1  = (__bf16*)(ws + ((size_t)8  << 20));          // 512 KB
    __bf16* wb2  = (__bf16*)(ws + ((size_t)9  << 20));          // ~29 MB

    const size_t shm1 = (size_t)1 * 64 * LDSROW * sizeof(__bf16); //  66.5 KB
    const size_t shm2 = (size_t)2 * 64 * LDSROW * sizeof(__bf16); // 133   KB

    // Zero output map + mask, then set mask diagonals.
    size_t n4 = (size_t)out_size / 4;
    zero_f4<<<8192, 256, 0, stream>>>((v4f*)out, n4);
    mask_kernel<<<1, 64, 0, stream>>>(mask);

    // Precision/layout conversion.
    cvt_x <<<2048, 256, 0, stream>>>(x,  actA);
    cvt_w1<<<1024, 256, 0, stream>>>(w1, wb1);
    cvt_w2<<<8192, 256, 0, stream>>>(w2, wb2);

    const __bf16* src = actA;
    __bf16*       dst = actB;
    auto swap = [&]() { const __bf16* t = dst; dst = (__bf16*)src; src = t; };

    // ---- level 0, order 0: 1x1 conv ----
    {
        int Lin = 64, Lout = 64;
        dim3 g(8, (64 * Lout + 127) / 128);
        conv_wmma<1><<<g, 256, shm1, stream>>>(src, dst, wb1, b1, x2d, Lin, Lout, 1, 0);
        swap();
    }
    int k2 = 0, L = 64;
    // ---- level 0, orders 1..15: k=2 convs, stride 1, offset = order ----
    for (int o = 1; o < 16; ++o) {
        int Lin = L, Lout = L - 1;
        dim3 g(8, (64 * Lout + 127) / 128);
        conv_wmma<2><<<g, 256, shm2, stream>>>(src, dst,
            wb2 + (size_t)k2 * 2 * 512 * 512, b2 + (size_t)k2 * 512,
            x2d, Lin, Lout, 1, o);
        ++k2; L = Lout; swap();
    }
    // ---- level 1, order 0: maxpool, stride 2, offset 17 ----
    {
        int Lin = L /*49*/, Lout = 24;
        int n = 64 * Lout * 512;
        maxpool_scatter<<<(n + 255) / 256, 256, 0, stream>>>(src, dst, x2d, Lin, Lout, 2, 17);
        L = Lout; swap();
    }
    // ---- level 1, orders 1..7 ----
    for (int o = 1; o < 8; ++o) {
        int Lin = L, Lout = L - 1;
        dim3 g(8, (64 * Lout + 127) / 128);
        conv_wmma<2><<<g, 256, shm2, stream>>>(src, dst,
            wb2 + (size_t)k2 * 2 * 512 * 512, b2 + (size_t)k2 * 512,
            x2d, Lin, Lout, 2, 17 + 2 * o);
        ++k2; L = Lout; swap();
    }
    // ---- level 2, order 0: maxpool, stride 4, offset 35 ----
    {
        int Lin = L /*17*/, Lout = 8;
        int n = 64 * Lout * 512;
        maxpool_scatter<<<(n + 255) / 256, 256, 0, stream>>>(src, dst, x2d, Lin, Lout, 4, 35);
        L = Lout; swap();
    }
    // ---- level 2, orders 1..7 ----
    for (int o = 1; o < 8; ++o) {
        int Lin = L, Lout = L - 1;
        dim3 g(8, (64 * Lout + 127) / 128);
        conv_wmma<2><<<g, 256, shm2, stream>>>(src, dst,
            wb2 + (size_t)k2 * 2 * 512 * 512, b2 + (size_t)k2 * 512,
            x2d, Lin, Lout, 4, 35 + 4 * o);
        ++k2; L = Lout; swap();
    }
    (void)n_in; (void)in_sizes; (void)ws_size;
}
